// Attention_70566312673329
// MI455X (gfx1250) — compile-verified, matrix operand order checked by reading, not run on hardware
//
#include <hip/hip_runtime.h>
#include <hip/hip_bf16.h>

#define S_LEN 4096
#define NHEAD 12
#define D_ATT 64
#define EMB   768

typedef __attribute__((ext_vector_type(16))) _Float16 v16h;
typedef __attribute__((ext_vector_type(8)))  float    v8f;
typedef __attribute__((ext_vector_type(4)))  unsigned v4u;
typedef __attribute__((ext_vector_type(8)))  unsigned v8u;

union Frag16 { v16h h; uint4 u[2]; };

__device__ __forceinline__ v16h load_frag(const _Float16* rowptr, int kb) {
    // elems 0..7  = row[kb .. kb+7]   (16B)
    // elems 8..15 = row[kb+16..kb+23] (16B)
    Frag16 f;
    f.u[0] = *(const uint4*)(rowptr + kb);
    f.u[1] = *(const uint4*)(rowptr + kb + 16);
    return f.h;
}

__device__ __forceinline__ v8f wmma_f16(v16h a, v16h b, v8f c) {
    return __builtin_amdgcn_wmma_f32_16x16x32_f16(
        /*neg_a=*/false, a, /*neg_b=*/false, b,
        /*c_mod=*/(short)0, c, /*reuse_a=*/false, /*reuse_b=*/false);
}

__device__ __forceinline__ unsigned lds_off(const void* p) {
    return (unsigned)(unsigned long long)p;   // LDS aperture: low 32 bits = LDS byte offset
}

// ---------------- Tensor Data Mover (TDM) helpers --------------------------
// D# group0: count=1 | lds_addr | global_addr(57b) | type=2 ("image")
__device__ __forceinline__ v4u tdm_g0(unsigned lds_addr, unsigned long long gaddr) {
    v4u g;
    g[0] = 1u;                                     // count=1, user descriptor
    g[1] = lds_addr;                               // bits 63:32
    g[2] = (unsigned)gaddr;                        // bits 95:64  = global_addr[31:0]
    g[3] = (unsigned)(gaddr >> 32) | (2u << 30);   // global_addr[56:32] | type=2
    return g;
}
// D# group1 for a 2D tile of 2-byte elements with LDS row padding.
// pad semantics (ISA 8.4): pad after 2^(pi+1) DWORDs stored, pad (pa+1) DWORDs.
__device__ __forceinline__ v8u tdm_g1(unsigned dim0, unsigned dim1, unsigned tile0,
                                      unsigned tile1, unsigned stride0,
                                      unsigned pad_interval, unsigned pad_amount) {
    v8u g;
    g[0] = (1u << 16) | (1u << 20) | (pad_interval << 22) | (pad_amount << 25);
    g[1] = (dim0 & 0xFFFFu) << 16;                 // tensor_dim0[15:0]
    g[2] = (dim0 >> 16) | ((dim1 & 0xFFFFu) << 16);// tensor_dim0[31:16] | tensor_dim1[15:0]
    g[3] = (dim1 >> 16) | (tile0 << 16);           // tensor_dim1[31:16] | tile_dim0
    g[4] = tile1;                                  // tile_dim1 (tile_dim2 = 0)
    g[5] = stride0;                                // tensor_dim0_stride[31:0]
    g[6] = 0;                                      // stride0 hi | tensor_dim1_stride lo
    g[7] = 0;
    return g;
}
__device__ __forceinline__ void tdm_load(v4u g0, v8u g1) {
    asm volatile("tensor_load_to_lds %0, %1" :: "s"(g0), "s"(g1) : "memory");
}
__device__ __forceinline__ void wait_tensor0() {
#if __has_builtin(__builtin_amdgcn_s_wait_tensorcnt)
    __builtin_amdgcn_s_wait_tensorcnt(0);
#else
    asm volatile("s_wait_tensorcnt 0x0" ::: "memory");
#endif
}

// ---------------- conversion kernels ----------------

__global__ void cvt_f32_f16(const float* __restrict__ in, _Float16* __restrict__ out, int n) {
    int i = blockIdx.x * blockDim.x + threadIdx.x;
    if (i < n) out[i] = (_Float16)in[i];
}

// W stored [in][out]; produce Wt[out][in] in f16 so B-fragments are row reads.
__global__ void transpose_cvt(const float* __restrict__ W, _Float16* __restrict__ Wt) {
    int i = blockIdx.x * blockDim.x + threadIdx.x;
    if (i < EMB * EMB) {
        int n = i / EMB, k = i % EMB;
        Wt[i] = (_Float16)W[(size_t)k * EMB + n];
    }
}

// ---------------- generic WMMA GEMM: C = A[M,K] * Bt[N,K]^T ----------------
// One wave computes a 32x32 output tile (2x2 sub-tiles, A/B fragments reused).
// mode 0: f32 output [M][N]                    (final projection -> d_out)
// mode 1: f16 head-split [H][S][64]            (Q, K)
// mode 2: f16 head-split transposed [H][64][S] (V)
__device__ __forceinline__ void store_tile(v8f c, int m0, int n0, int r, int mhi,
                                           _Float16* __restrict__ outH, float* __restrict__ outF,
                                           int M, int N, int mode) {
#pragma unroll
    for (int j = 0; j < 8; ++j) {
        int m = m0 + j + mhi;
        int n = n0 + r;
        if (mode == 0) {
            outF[(size_t)m * N + n] = c[j];
        } else if (mode == 1) {
            int h = n >> 6, d = n & 63;
            outH[((size_t)h * M + m) * D_ATT + d] = (_Float16)c[j];
        } else {
            int h = n >> 6, d = n & 63;
            outH[((size_t)h * D_ATT + d) * M + m] = (_Float16)c[j];
        }
    }
}

__global__ void gemm_wmma(const _Float16* __restrict__ A, const _Float16* __restrict__ Bt,
                          _Float16* __restrict__ outH, float* __restrict__ outF,
                          int M, int N, int K, int mode) {
    int lane = threadIdx.x & 31;
    int wave = threadIdx.x >> 5;
    int tile = blockIdx.x * (blockDim.x >> 5) + wave;
    int tilesN = N >> 5;
    int total = (M >> 5) * tilesN;
    if (tile >= total) return;                 // wave-uniform: EXEC all-ones inside

    int m0 = (tile / tilesN) << 5;
    int n0 = (tile % tilesN) << 5;
    int r   = lane & 15;
    int kb  = (lane >> 4) << 3;
    int mhi = (lane >> 4) << 3;

    const _Float16* arow0 = A  + (size_t)(m0 + r) * K;
    const _Float16* arow1 = arow0 + (size_t)16 * K;
    const _Float16* brow0 = Bt + (size_t)(n0 + r) * K;
    const _Float16* brow1 = brow0 + (size_t)16 * K;

    v8f c00 = {}, c01 = {}, c10 = {}, c11 = {};
#pragma unroll 2
    for (int k0 = 0; k0 < K; k0 += 32) {
        v16h a0 = load_frag(arow0 + k0, kb);
        v16h a1 = load_frag(arow1 + k0, kb);
        v16h b0 = load_frag(brow0 + k0, kb);
        v16h b1 = load_frag(brow1 + k0, kb);
        c00 = wmma_f16(a0, b0, c00);
        c01 = wmma_f16(a0, b1, c01);
        c10 = wmma_f16(a1, b0, c10);
        c11 = wmma_f16(a1, b1, c11);
    }
    store_tile(c00, m0,      n0,      r, mhi, outH, outF, M, N, mode);
    store_tile(c01, m0,      n0 + 16, r, mhi, outH, outF, M, N, mode);
    store_tile(c10, m0 + 16, n0,      r, mhi, outH, outF, M, N, mode);
    store_tile(c11, m0 + 16, n0 + 16, r, mhi, outH, outF, M, N, mode);
}

// ---------------- row norms ||q||^2, ||k||^2 over head dim ----------------
__global__ void row_norms(const _Float16* __restrict__ Qh, const _Float16* __restrict__ Kh,
                          float* __restrict__ qn, float* __restrict__ kn) {
    int i = blockIdx.x * blockDim.x + threadIdx.x;   // over H*S rows
    if (i >= NHEAD * S_LEN) return;
    const _Float16* q = Qh + (size_t)i * D_ATT;
    const _Float16* k = Kh + (size_t)i * D_ATT;
    float sq = 0.f, sk = 0.f;
#pragma unroll 8
    for (int d = 0; d < D_ATT; ++d) {
        float a = (float)q[d]; sq += a * a;
        float b = (float)k[d]; sk += b * b;
    }
    qn[i] = sq; kn[i] = sk;
}

// ---------------- fused RBF attention -------------------------------------
// Block = 4 waves, one head, 128 query rows (32 per wave = 2 m-tiles).
// K (32x64) and V (64x32) tiles are staged into double-buffered LDS by the
// Tensor Data Mover (wave 0 issues tensor_load_to_lds; TENSORcnt + barrier
// publishes). Staging of tile i+1 overlaps compute of tile i; one barrier
// per key step. RBF attention has no softmax denominator -> plain accumulate.
__global__ void __launch_bounds__(128)
rbf_attn(const _Float16* __restrict__ Qh, const _Float16* __restrict__ Kh,
         const _Float16* __restrict__ Vt, const float* __restrict__ qn,
         const float* __restrict__ kn, const float* __restrict__ gamma,
         _Float16* __restrict__ outH) {
    __shared__ __align__(16) _Float16 sK[2][32][72];     // keys x d, 144B padded rows
    __shared__ __align__(16) _Float16 sV[2][64][40];     // d x keys,  80B padded rows
    __shared__ __align__(16) _Float16 sS[4][2][16][40];  // per-wave, per-mtile score slabs

    int tid  = threadIdx.x;
    int lane = tid & 31;
    int wv   = tid >> 5;
    int h    = blockIdx.y;
    int r    = lane & 15;
    int kb   = (lane >> 4) << 3;
    int mhi  = kb;
    int sBlk = blockIdx.x * 128 + wv * 32;

    float g = gamma[h] * 0.125f;                         // SCALE = 1/sqrt(64)

    // Loop-invariant Q fragments + q-norms for both m-tiles
    v16h aq0[2], aq1[2];
    float qr[2][8];
#pragma unroll
    for (int mt = 0; mt < 2; ++mt) {
        const _Float16* qrow = Qh + ((size_t)h * S_LEN + sBlk + mt * 16 + r) * D_ATT;
        aq0[mt] = load_frag(qrow, kb);
        aq1[mt] = load_frag(qrow + 32, kb);
#pragma unroll
        for (int j = 0; j < 8; ++j)
            qr[mt][j] = qn[(size_t)h * S_LEN + sBlk + mt * 16 + j + mhi];
    }

    v8f acc[2][4];
#pragma unroll
    for (int mt = 0; mt < 2; ++mt)
#pragma unroll
        for (int dt = 0; dt < 4; ++dt) acc[mt][dt] = v8f{};

    // TDM descriptors (wave-uniform).  K tile: 64-wide rows (128B = 16x8B ->
    // pad_interval=4), pad 4 DWORDs -> 72-half stride.  V tile: 32-wide rows
    // (64B = 8x8B -> pad_interval=3), pad 4 DWORDs -> 40-half stride.
    const v8u g1K = tdm_g1(/*dim0=*/D_ATT, /*dim1=*/32, /*tile0=*/D_ATT, /*tile1=*/32,
                           /*stride0=*/D_ATT, /*pi=*/4, /*pa=*/3);
    const v8u g1V = tdm_g1(/*dim0=*/S_LEN, /*dim1=*/D_ATT, /*tile0=*/32, /*tile1=*/D_ATT,
                           /*stride0=*/S_LEN, /*pi=*/3, /*pa=*/3);
    unsigned long long kga = (unsigned long long)(Kh + (size_t)h * S_LEN * D_ATT);
    unsigned long long vga = (unsigned long long)(Vt + (size_t)h * D_ATT * S_LEN);
    unsigned lK[2] = { lds_off(&sK[0][0][0]), lds_off(&sK[1][0][0]) };
    unsigned lV[2] = { lds_off(&sV[0][0][0]), lds_off(&sV[1][0][0]) };

    if (wv == 0) {                                        // prime buffer 0 (t0 = 0)
        tdm_load(tdm_g0(lK[0], kga), g1K);
        tdm_load(tdm_g0(lV[0], vga), g1V);
    }

    for (int t0 = 0; t0 < S_LEN; t0 += 32) {
        int b = (t0 >> 5) & 1;
        if (wv == 0) wait_tensor0();                      // buffer b filled
        __syncthreads();                                  // publish to all waves
        if (wv == 0 && t0 + 32 < S_LEN) {                 // stage next tile into b^1
            tdm_load(tdm_g0(lK[b ^ 1], kga + (size_t)(t0 + 32) * D_ATT * 2), g1K);
            tdm_load(tdm_g0(lV[b ^ 1], vga + (size_t)(t0 + 32) * 2), g1V);
        }
        // L2 prefetch two tiles ahead (global_prefetch_b8)
        if (t0 + 64 < S_LEN) {
            __builtin_prefetch(Kh + ((size_t)h * S_LEN + t0 + 64 + (tid & 31)) * D_ATT +
                               ((tid >> 5) << 5), 0, 0);
            __builtin_prefetch(Vt + ((size_t)h * D_ATT + (tid >> 1)) * S_LEN + t0 + 64 +
                               ((tid & 1) << 4), 0, 0);
        }

        // --- QK^T -> dist -> exp scores (both m-tiles, both 16-key n-tiles) ---
#pragma unroll
        for (int mt = 0; mt < 2; ++mt) {
#pragma unroll
            for (int nt = 0; nt < 2; ++nt) {
                const _Float16* krow_l = &sK[b][nt * 16 + r][0];
                v16h b0 = load_frag(krow_l, kb);
                v16h b1 = load_frag(krow_l + 32, kb);
                v8f c = {};
                c = wmma_f16(aq0[mt], b0, c);
                c = wmma_f16(aq1[mt], b1, c);
                float knv = kn[(size_t)h * S_LEN + t0 + nt * 16 + r];
#pragma unroll
                for (int j = 0; j < 8; ++j) {
                    float dist = qr[mt][j] - 2.0f * c[j] + knv;
                    dist = fmaxf(dist, 0.0f);
                    float sc = __expf(-g * dist);
                    sS[wv][mt][j + mhi][nt * 16 + r] = (_Float16)sc;  // [m][t] slab
                }
            }
        }

        // --- scores @ V: re-read slab in A-layout (wave-private; DS in-order) ---
#pragma unroll
        for (int mt = 0; mt < 2; ++mt) {
            Frag16 as;
            as.u[0] = *(const uint4*)&sS[wv][mt][r][kb];
            as.u[1] = *(const uint4*)&sS[wv][mt][r][kb + 16];
#pragma unroll
            for (int dt = 0; dt < 4; ++dt) {
                v16h bv = load_frag(&sV[b][dt * 16 + r][0], kb);
                acc[mt][dt] = wmma_f16(as.h, bv, acc[mt][dt]);
            }
        }
        // no trailing barrier: buffer b is re-staged only after next top barrier
    }

#pragma unroll
    for (int mt = 0; mt < 2; ++mt)
#pragma unroll
        for (int dt = 0; dt < 4; ++dt)
#pragma unroll
            for (int j = 0; j < 8; ++j)
                outH[(size_t)(sBlk + mt * 16 + j + mhi) * EMB + h * D_ATT + dt * 16 + r] =
                    (_Float16)acc[mt][dt][j];
}

// ---------------- launcher ----------------

extern "C" void kernel_launch(void* const* d_in, const int* in_sizes, int n_in,
                              void* d_out, int out_size, void* d_ws, size_t ws_size,
                              hipStream_t stream) {
    const float* x     = (const float*)d_in[0];
    const float* Wq    = (const float*)d_in[1];
    const float* Wk    = (const float*)d_in[2];
    const float* Wv    = (const float*)d_in[3];
    const float* Wo    = (const float*)d_in[4];
    const float* gamma = (const float*)d_in[5];

    char* ws = (char*)d_ws;
    const size_t SZ_XH = (size_t)S_LEN * EMB * sizeof(_Float16);   // 6.0 MB
    const size_t SZ_W  = (size_t)EMB * EMB * sizeof(_Float16);     // 1.2 MB
    const size_t SZ_N  = (size_t)NHEAD * S_LEN * sizeof(float);    // 192 KB

    _Float16* xh   = (_Float16*)(ws);
    _Float16* Wqt  = (_Float16*)(ws + SZ_XH);
    _Float16* Wkt  = (_Float16*)(ws + SZ_XH + SZ_W);
    _Float16* Wvt  = (_Float16*)(ws + SZ_XH + 2 * SZ_W);
    _Float16* Wot  = (_Float16*)(ws + SZ_XH + 3 * SZ_W);
    _Float16* Qh   = (_Float16*)(ws + SZ_XH + 4 * SZ_W);
    _Float16* Kh   = (_Float16*)(ws + 2 * SZ_XH + 4 * SZ_W);
    _Float16* Vt   = (_Float16*)(ws + 3 * SZ_XH + 4 * SZ_W);
    _Float16* outh = (_Float16*)(ws + 4 * SZ_XH + 4 * SZ_W);
    float*    qn   = (float*)   (ws + 5 * SZ_XH + 4 * SZ_W);
    float*    kn   = (float*)   (ws + 5 * SZ_XH + 4 * SZ_W + SZ_N);

    const int nX = S_LEN * EMB;      // 3,145,728
    const int nW = EMB * EMB;        // 589,824
    cvt_f32_f16  <<<(nX + 255) / 256, 256, 0, stream>>>(x, xh, nX);
    transpose_cvt<<<(nW + 255) / 256, 256, 0, stream>>>(Wq, Wqt);
    transpose_cvt<<<(nW + 255) / 256, 256, 0, stream>>>(Wk, Wkt);
    transpose_cvt<<<(nW + 255) / 256, 256, 0, stream>>>(Wv, Wvt);
    transpose_cvt<<<(nW + 255) / 256, 256, 0, stream>>>(Wo, Wot);

    const int tiles  = (S_LEN / 32) * (EMB / 32);     // 3072 wave-tiles (32x32 each)
    const int blocks = (tiles + 7) / 8;               // 8 waves / 256-thread block
    gemm_wmma<<<blocks, 256, 0, stream>>>(xh, Wqt, Qh, nullptr, S_LEN, EMB, EMB, 1);
    gemm_wmma<<<blocks, 256, 0, stream>>>(xh, Wkt, Kh, nullptr, S_LEN, EMB, EMB, 1);
    gemm_wmma<<<blocks, 256, 0, stream>>>(xh, Wvt, Vt, nullptr, S_LEN, EMB, EMB, 2);

    const int nRows = NHEAD * S_LEN;
    row_norms<<<(nRows + 255) / 256, 256, 0, stream>>>(Qh, Kh, qn, kn);

    rbf_attn<<<dim3(S_LEN / 128, NHEAD), 128, 0, stream>>>(Qh, Kh, Vt, qn, kn, gamma, outh);

    gemm_wmma<<<blocks, 256, 0, stream>>>(outh, Wot, nullptr, (float*)d_out, S_LEN, EMB, EMB, 0);
}
